// GraphTransformerLayer_22359599743562
// MI455X (gfx1250) — compile-verified, matrix operand order checked by reading, not run on hardware
//
#include <hip/hip_runtime.h>
#include <hip/hip_bf16.h>

#define NN   4096
#define DD   256
#define HH   8
#define HDIM 32
#define EE   131072
#define DFF  1024
#define MAXNB 512

typedef __attribute__((ext_vector_type(16))) _Float16 v16h;
typedef __attribute__((ext_vector_type(8)))  _Float16 v8h;
typedef __attribute__((ext_vector_type(8)))  float    v8f;

// ---------------- f32 -> f16 convert ----------------
__global__ void k_cvt(const float* __restrict__ s, _Float16* __restrict__ d, int n) {
    int i = blockIdx.x * 256 + threadIdx.x;
    if (i < n) d[i] = (_Float16)s[i];
}

// ---- f32 [K,N] -> f16 transposed [N,K] (coalesced writes; strided reads hit L2) ----
__global__ void k_cvt_t(const float* __restrict__ s, _Float16* __restrict__ d, int K, int N) {
    int i = blockIdx.x * 256 + threadIdx.x;
    if (i < K * N) {
        int n = i / K, k = i % K;
        d[i] = (_Float16)s[(size_t)k * N + n];
    }
}

// ---------------- CSR build ----------------
__global__ void k_edge_count(const long long* __restrict__ ei, int* __restrict__ counts) {
    int e = blockIdx.x * 256 + threadIdx.x;
    if (e < EE) atomicAdd(&counts[(int)ei[e]], 1);
}

__global__ void k_scan(const int* __restrict__ counts, int* __restrict__ rowptr,
                       int* __restrict__ cursor) {
    if (threadIdx.x == 0 && blockIdx.x == 0) {
        int acc = 0;
        for (int i = 0; i < NN; ++i) { rowptr[i] = acc; cursor[i] = acc; acc += counts[i]; }
        rowptr[NN] = acc;
    }
}

__global__ void k_edge_fill(const long long* __restrict__ ei, int* __restrict__ cursor,
                            int* __restrict__ cols, int* __restrict__ eidx) {
    int e = blockIdx.x * 256 + threadIdx.x;
    if (e < EE) {
        int r = (int)ei[e];
        int pos = atomicAdd(&cursor[r], 1);
        cols[pos] = (int)ei[EE + e];
        eidx[pos] = e;
    }
}

// ---------------- edge bias: eb[e,h] = edge_attr[e,:] @ we[:,h] + be[h] ----------------
__global__ void k_edge_bias(const float* __restrict__ ea, const float* __restrict__ we,
                            const float* __restrict__ be, float* __restrict__ eb) {
    __shared__ float swe[DD * HH];  // 8KB
    int tid = threadIdx.x;
    for (int i = tid; i < DD * HH; i += 256) swe[i] = we[i];
    __syncthreads();
    int wid = tid >> 5, lane = tid & 31;
    int e = blockIdx.x * 8 + wid;
    if (e >= EE) return;
    float p[HH];
#pragma unroll
    for (int h = 0; h < HH; ++h) p[h] = 0.f;
#pragma unroll
    for (int i = 0; i < 8; ++i) {
        int k = lane + i * 32;
        float v = ea[(size_t)e * DD + k];
#pragma unroll
        for (int h = 0; h < HH; ++h) p[h] += v * swe[k * HH + h];
    }
#pragma unroll
    for (int h = 0; h < HH; ++h)
        for (int off = 16; off > 0; off >>= 1) p[h] += __shfl_xor(p[h], off, 32);
    if (lane == 0) {
#pragma unroll
        for (int h = 0; h < HH; ++h) eb[(size_t)e * HH + h] = p[h] + be[h];
    }
}

// ---------------- WMMA tiled GEMM: C = A(f16)[M,K] @ Bt(f16)[N,K]^T + bias, opt GELU ----
// One wave per 16x16 tile. A and Bt fragments are each two contiguous 16B vector loads
// per k-step (CDNA5 16-bit fragment layout: lane half owns K +0/+8, VGPRs 0-3 = K 0..7,
// VGPRs 4-7 = K 16..23).
template <int K>
__global__ __launch_bounds__(256) void k_gemm_wmma(
        const _Float16* __restrict__ A, const _Float16* __restrict__ Bt,
        const float* __restrict__ bias, float* __restrict__ Cf,
        _Float16* __restrict__ Ch, int M, int N, int act) {
    int wid = threadIdx.x >> 5, lane = threadIdx.x & 31;
    int ntilesN = N >> 4;
    int tiles = (M >> 4) * ntilesN;
    int tile = blockIdx.x * 8 + wid;
    if (tile >= tiles) return;
    int tm = tile / ntilesN, tn = tile % ntilesN;
    int r = lane & 15, hf = lane >> 4;
    const int m = (tm << 4) + r;   // A row for this lane
    const int n = (tn << 4) + r;   // B/C col for this lane
    const _Float16* arow = A + (size_t)m * K;
    const _Float16* brow = Bt + (size_t)n * K;
    v8f acc = {};
#pragma unroll 4
    for (int k0 = 0; k0 < K; k0 += 32) {
        v8h alo = *(const v8h*)(arow + k0 + hf * 8);
        v8h ahi = *(const v8h*)(arow + k0 + 16 + hf * 8);
        v8h blo = *(const v8h*)(brow + k0 + hf * 8);
        v8h bhi = *(const v8h*)(brow + k0 + 16 + hf * 8);
        v16h a = __builtin_shufflevector(alo, ahi, 0, 1, 2, 3, 4, 5, 6, 7,
                                         8, 9, 10, 11, 12, 13, 14, 15);
        v16h b = __builtin_shufflevector(blo, bhi, 0, 1, 2, 3, 4, 5, 6, 7,
                                         8, 9, 10, 11, 12, 13, 14, 15);
        acc = __builtin_amdgcn_wmma_f32_16x16x32_f16(false, a, false, b, (short)0, acc,
                                                     false, false);
    }
    float bcol = bias[n];
#pragma unroll
    for (int v = 0; v < 8; ++v) {
        int row = (tm << 4) + hf * 8 + v;
        float val = acc[v] + bcol;
        if (act) val = 0.5f * val * (1.0f + erff(val * 0.70710678118f));
        if (Cf) Cf[(size_t)row * N + n] = val;
        if (Ch) Ch[(size_t)row * N + n] = (_Float16)val;
    }
}

// ---------------- sparse masked attention: one block per row ----------------
__global__ void k_attention(const float* __restrict__ Q, const float* __restrict__ Km,
                            const float* __restrict__ V, const unsigned char* __restrict__ adj,
                            const int* __restrict__ rowptr, const int* __restrict__ cols,
                            const int* __restrict__ eidx, const float* __restrict__ eb,
                            float* __restrict__ ctxf, _Float16* __restrict__ ctxh) {
    int row = blockIdx.x, tid = threadIdx.x;
    int wid = tid >> 5, lane = tid & 31;
    __shared__ float qs[DD];
    __shared__ int   nbr[MAXNB];
    __shared__ float lg[HH][MAXNB];
    __shared__ int   sc[256];
    __shared__ int   totNB;

    qs[tid] = Q[(size_t)row * DD + tid];

    // deterministic ordered compaction of adjacency row (16 contiguous cols per thread)
    const unsigned char* arow = adj + (size_t)row * NN;
    int base = tid * 16, lc = 0;
#pragma unroll
    for (int i = 0; i < 16; ++i) lc += arow[base + i] ? 1 : 0;
    sc[tid] = lc;
    __syncthreads();
    for (int s = 1; s < 256; s <<= 1) {
        int v = (tid >= s) ? sc[tid - s] : 0;
        __syncthreads();
        sc[tid] += v;
        __syncthreads();
    }
    int start = sc[tid] - lc;
    if (tid == 255) totNB = min(sc[255], MAXNB);
    for (int i = 0; i < 16; ++i)
        if (arow[base + i]) { if (start < MAXNB) nbr[start] = base + i; ++start; }
    __syncthreads();
    int NB = totNB;

    // logits: wave per neighbor-stride, lane-parallel over head dim, shuffle reduce
    for (int j = wid; j < NB; j += 8) {
        int mcol = nbr[j];
        const float* krow = Km + (size_t)mcol * DD;
#pragma unroll
        for (int h = 0; h < HH; ++h) {
            float v = krow[h * HDIM + lane] * qs[h * HDIM + lane];
            for (int off = 16; off > 0; off >>= 1) v += __shfl_xor(v, off, 32);
            if (lane == 0) lg[h][j] = v * 0.17677669529663687f;  // 1/sqrt(32)
        }
    }
    __syncthreads();

    // add edge bias where the edge lands on an adjacent (kept) entry
    for (int p = rowptr[row] + tid; p < rowptr[row + 1]; p += 256) {
        int c = cols[p], e = eidx[p];
        for (int j = 0; j < NB; ++j) {
            if (nbr[j] == c) {
#pragma unroll
                for (int h = 0; h < HH; ++h) atomicAdd(&lg[h][j], eb[(size_t)e * HH + h]);
                break;
            }
        }
    }
    __syncthreads();

    // softmax per head: one wave per head
    {
        int h = wid;
        float mx = -1e30f;
        for (int j = lane; j < NB; j += 32) mx = fmaxf(mx, lg[h][j]);
        for (int off = 16; off > 0; off >>= 1) mx = fmaxf(mx, __shfl_xor(mx, off, 32));
        float sum = 0.f;
        for (int j = lane; j < NB; j += 32) {
            float ev = expf(lg[h][j] - mx);
            lg[h][j] = ev;
            sum += ev;
        }
        for (int off = 16; off > 0; off >>= 1) sum += __shfl_xor(sum, off, 32);
        float inv = 1.0f / sum;
        for (int j = lane; j < NB; j += 32) lg[h][j] *= inv;
    }
    __syncthreads();

    // ctx[row, h*32+d] = sum_j attn[j,h] * V[nbr[j], h*32+d]
    {
        int h = wid, d = lane;
        float acc = 0.f;
        for (int j = 0; j < NB; ++j)
            acc += lg[h][j] * V[(size_t)nbr[j] * DD + h * HDIM + d];
        size_t o = (size_t)row * DD + h * HDIM + d;
        ctxf[o] = acc;
        ctxh[o] = (_Float16)acc;
    }
}

// ---------------- residual + LayerNorm ----------------
__global__ void k_add_ln(const float* __restrict__ a, const float* __restrict__ b,
                         const float* __restrict__ g, const float* __restrict__ be,
                         float* __restrict__ of, _Float16* __restrict__ oh) {
    int row = blockIdx.x, tid = threadIdx.x;
    float v = a[(size_t)row * DD + tid] + b[(size_t)row * DD + tid];
    __shared__ float s1[256], s2[256];
    s1[tid] = v;
    s2[tid] = v * v;
    __syncthreads();
    for (int s = 128; s > 0; s >>= 1) {
        if (tid < s) { s1[tid] += s1[tid + s]; s2[tid] += s2[tid + s]; }
        __syncthreads();
    }
    float mu = s1[0] * (1.0f / DD);
    float var = s2[0] * (1.0f / DD) - mu * mu;
    float y = (v - mu) * rsqrtf(var + 1e-5f) * g[tid] + be[tid];
    of[(size_t)row * DD + tid] = y;
    if (oh) oh[(size_t)row * DD + tid] = (_Float16)y;
}

// ---------------- host orchestration ----------------
extern "C" void kernel_launch(void* const* d_in, const int* in_sizes, int n_in,
                              void* d_out, int out_size, void* d_ws, size_t ws_size,
                              hipStream_t stream) {
    const float*         x    = (const float*)d_in[0];
    const unsigned char* adj  = (const unsigned char*)d_in[1];
    const long long*     ei   = (const long long*)d_in[2];
    const float*         ea   = (const float*)d_in[3];
    const float *wq = (const float*)d_in[4],  *bq = (const float*)d_in[5];
    const float *wk = (const float*)d_in[6],  *bk = (const float*)d_in[7];
    const float *wv = (const float*)d_in[8],  *bv = (const float*)d_in[9];
    const float *wo = (const float*)d_in[10], *bo = (const float*)d_in[11];
    const float *we = (const float*)d_in[12], *be = (const float*)d_in[13];
    const float *w1 = (const float*)d_in[14], *b1 = (const float*)d_in[15];
    const float *w2 = (const float*)d_in[16], *b2 = (const float*)d_in[17];
    const float *g1 = (const float*)d_in[18], *be1 = (const float*)d_in[19];
    const float *g2 = (const float*)d_in[20], *be2 = (const float*)d_in[21];
    float* out = (float*)d_out;

    char* wsb = (char*)d_ws;
    size_t off = 0;
    auto alloc = [&](size_t bytes) -> void* {
        void* p = wsb + off;
        off = (off + bytes + 255) & ~(size_t)255;
        return p;
    };
    _Float16* xh   = (_Float16*)alloc(2ull * NN * DD);
    _Float16* wqt  = (_Float16*)alloc(2ull * DD * DD);   // transposed f16 weights [N,K]
    _Float16* wkt  = (_Float16*)alloc(2ull * DD * DD);
    _Float16* wvt  = (_Float16*)alloc(2ull * DD * DD);
    _Float16* wot  = (_Float16*)alloc(2ull * DD * DD);
    _Float16* w1t  = (_Float16*)alloc(2ull * DD * DFF);  // [DFF, DD]
    _Float16* w2t  = (_Float16*)alloc(2ull * DFF * DD);  // [DD, DFF]
    float*    Qf   = (float*)alloc(4ull * NN * DD);
    float*    Kf   = (float*)alloc(4ull * NN * DD);
    float*    Vf   = (float*)alloc(4ull * NN * DD);
    float*    ctxf = (float*)alloc(4ull * NN * DD);
    _Float16* ctxh = (_Float16*)alloc(2ull * NN * DD);
    float*    aof  = (float*)alloc(4ull * NN * DD);
    float*    x1f  = (float*)alloc(4ull * NN * DD);
    _Float16* x1h  = (_Float16*)alloc(2ull * NN * DD);
    _Float16* hidh = (_Float16*)alloc(2ull * NN * DFF);
    float*    fff  = (float*)alloc(4ull * NN * DD);
    float*    eb   = (float*)alloc(4ull * EE * HH);
    int*      cnts = (int*)alloc(4ull * NN);
    int*      rptr = (int*)alloc(4ull * (NN + 1));
    int*      curs = (int*)alloc(4ull * NN);
    int*      ecol = (int*)alloc(4ull * EE);
    int*      eidd = (int*)alloc(4ull * EE);

    // f16 conversions (weights transposed so WMMA B fragments are contiguous b128 loads)
    k_cvt<<<(NN * DD) / 256, 256, 0, stream>>>(x, xh, NN * DD);
    k_cvt_t<<<(DD * DD) / 256, 256, 0, stream>>>(wq, wqt, DD, DD);
    k_cvt_t<<<(DD * DD) / 256, 256, 0, stream>>>(wk, wkt, DD, DD);
    k_cvt_t<<<(DD * DD) / 256, 256, 0, stream>>>(wv, wvt, DD, DD);
    k_cvt_t<<<(DD * DD) / 256, 256, 0, stream>>>(wo, wot, DD, DD);
    k_cvt_t<<<(DD * DFF) / 256, 256, 0, stream>>>(w1, w1t, DD, DFF);   // -> [DFF, DD]
    k_cvt_t<<<(DFF * DD) / 256, 256, 0, stream>>>(w2, w2t, DFF, DD);   // -> [DD, DFF]

    // CSR of edges grouped by source row (rebuilt every call: deterministic)
    hipMemsetAsync(cnts, 0, 4ull * NN, stream);
    k_edge_count<<<EE / 256, 256, 0, stream>>>(ei, cnts);
    k_scan<<<1, 32, 0, stream>>>(cnts, rptr, curs);
    k_edge_fill<<<EE / 256, 256, 0, stream>>>(ei, curs, ecol, eidd);
    k_edge_bias<<<EE / 8, 256, 0, stream>>>(ea, we, be, eb);

    // QKV projections (WMMA, K=256)
    int tilesQ = (NN / 16) * (DD / 16);
    k_gemm_wmma<DD><<<tilesQ / 8, 256, 0, stream>>>(xh, wqt, bq, Qf, nullptr, NN, DD, 0);
    k_gemm_wmma<DD><<<tilesQ / 8, 256, 0, stream>>>(xh, wkt, bk, Kf, nullptr, NN, DD, 0);
    k_gemm_wmma<DD><<<tilesQ / 8, 256, 0, stream>>>(xh, wvt, bv, Vf, nullptr, NN, DD, 0);

    // sparse masked attention
    k_attention<<<NN, 256, 0, stream>>>(Qf, Kf, Vf, adj, rptr, ecol, eidd, eb, ctxf, ctxh);

    // output projection + residual LN
    k_gemm_wmma<DD><<<tilesQ / 8, 256, 0, stream>>>(ctxh, wot, bo, aof, nullptr, NN, DD, 0);
    k_add_ln<<<NN, 256, 0, stream>>>(x, aof, g1, be1, x1f, x1h);

    // FFN (WMMA): GELU in epilogue of GEMM1
    int tilesF1 = (NN / 16) * (DFF / 16);
    k_gemm_wmma<DD><<<tilesF1 / 8, 256, 0, stream>>>(x1h, w1t, b1, nullptr, hidh, NN, DFF, 1);
    k_gemm_wmma<DFF><<<tilesQ / 8, 256, 0, stream>>>(hidh, w2t, b2, fff, nullptr, NN, DD, 0);

    // final residual LN -> output
    k_add_ln<<<NN, 256, 0, stream>>>(x1f, fff, g2, be2, out, nullptr);

    (void)in_sizes; (void)n_in; (void)out_size; (void)ws_size;
}
